// LstmAtt_S_41257455845901
// MI455X (gfx1250) — matrix operand order repaired and checked by reading, not compiled, not run on hardware
//
#include <hip/hip_runtime.h>
#include <hip/hip_bf16.h>
#include <stdint.h>

typedef __attribute__((ext_vector_type(16))) __bf16 v16bf;
typedef __attribute__((ext_vector_type(8)))  float  v8f;
typedef unsigned short u16;

#define BT   16     // batch tile per workgroup
#define HH   768
#define LL   128
#define DIN  130
#define K3D  390    // 3*D_IN
#define K3DP 416    // padded to 13*32
#define KBH  24     // 768/32 k-blocks
#define KBV  13     // 416/32 k-blocks
#define TPG  48     // 768/16 N-tiles per gate
#define FEAT 49
#define PROJ 40
#define NTHR 768    // 24 waves
#define JPW  2      // j-tiles (16 h-columns each) per wave

// float -> bf16 bits (round to nearest even)
__device__ __forceinline__ u16 f2bf(float f) {
  union { float f; unsigned u; } x; x.f = f;
  unsigned u = x.u;
  return (u16)((u + 0x7FFFu + ((u >> 16) & 1u)) >> 16);
}

// Optimization barrier that PRESERVES pointer provenance (so addrspace
// inference still yields global_load saddr-form, not flat_load): add an
// opaque zero offset held in an SGPR. Re-materialized inside the step loop,
// it defeats cross-step LICM/CSE of the loop-invariant weight streams.
template <typename T>
__device__ __forceinline__ const T* opaque(const T* p) {
  unsigned long long z = 0;
  asm volatile("" : "+s"(z));
  return p + z;
}

// element offset (in u16 units) of h[m][k] inside the 16-bit A-fragment layout:
// per 32-wide k-block: 32 lanes x 16 packed bf16; lane = m + 16*((k>>4)&1),
// within-lane elem = k&15   (contiguous k order, matches B packing)
__device__ __forceinline__ int a_elem(int m, int k) {
  int lane = m + (((k >> 4) & 1) << 4);
  int e = k & 15;
  return ((k >> 5) << 9) + (lane << 4) + e;
}

__device__ __forceinline__ float sigmoidf_(float x) { return 1.0f / (1.0f + __expf(-x)); }

// ---------------- weight repack kernels (run once per launch) ----------------
// B-fragment layout for 16-bit 32x16 B tiles: lane = (n&15) + 16*((k>>4)&1),
// within-lane elem = k&15. Fragment-major: dst[(ntile*KB + kb)*512 + lane*16 + e]

__global__ void pack_watt(const float* __restrict__ W, u16* __restrict__ dst) {
  int idx = blockIdx.x * blockDim.x + threadIdx.x;   // over 768*384
  if (idx >= HH * 384) return;
  int k = idx / 384, n = idx % 384;                  // B[k][n] = W_att[n/128][k][n%128]
  float v = W[(n >> 7) * (HH * LL) + k * LL + (n & 127)];
  int lane = (n & 15) + (((k >> 4) & 1) << 4);
  dst[(((n >> 4) * KBH + (k >> 5)) << 9) + (lane << 4) + (k & 15)] = f2bf(v);
}

__global__ void pack_whh(const float* __restrict__ W, u16* __restrict__ dst) {
  int idx = blockIdx.x * blockDim.x + threadIdx.x;   // over 3072*768
  if (idx >= 3072 * HH) return;
  int n = idx / HH, k = idx - n * HH;                // B[k][n] = W_hh[n][k]
  float v = W[idx];
  int lane = (n & 15) + (((k >> 4) & 1) << 4);
  dst[(((n >> 4) * KBH + (k >> 5)) << 9) + (lane << 4) + (k & 15)] = f2bf(v);
}

__global__ void pack_wih(const float* __restrict__ W, u16* __restrict__ dst) {
  int idx = blockIdx.x * blockDim.x + threadIdx.x;   // over 3072*416 (padded K)
  if (idx >= 3072 * K3DP) return;
  int n = idx / K3DP, k = idx - n * K3DP;
  float v = (k < K3D) ? W[n * K3D + k] : 0.0f;       // zero-pad K 390..415
  int lane = (n & 15) + (((k >> 4) & 1) << 4);
  dst[(((n >> 4) * KBV + (k >> 5)) << 9) + (lane << 4) + (k & 15)] = f2bf(v);
}

__global__ void add_bias(const float* __restrict__ a, const float* __restrict__ b,
                         float* __restrict__ c) {
  int i = blockIdx.x * blockDim.x + threadIdx.x;
  if (i < 4 * HH) c[i] = a[i] + b[i];
}

// ---------------- main fused kernel: 64 WGs x 768 threads (24 waves) --------
__global__ void __launch_bounds__(NTHR, 1)
lstm_att_main(const float* __restrict__ X, const float* __restrict__ Fe,
              const float* __restrict__ b_att,
              const u16* __restrict__ WattF, const u16* __restrict__ WhhF,
              const u16* __restrict__ WihF, const float* __restrict__ bcomb,
              const float* __restrict__ Wd1, const float* __restrict__ bd1,
              const float* __restrict__ Wr,  const float* __restrict__ br,
              const float* __restrict__ Wa,  const float* __restrict__ Wd2,
              const float* __restrict__ bd2, float* __restrict__ out)
{
  __shared__ __align__(32) char smem[63488];
  u16*   hA  = (u16*)smem;                 // h in A-frag layout: 24*512 u16 = 24576 B
  u16*   vA  = (u16*)(smem + 24576);       // v in A-frag layout: 13*512 u16 = 13312 B
  float* att = (float*)(smem + 37888);     // logits/alpha [16][384] f32 = 24576 B

  const int tid  = threadIdx.x;
  const int wave = tid >> 5;
  const int lane = tid & 31;
  const int b0   = blockIdx.x * BT;
  const int jbase = wave * JPW;            // each wave owns 2 h-column tiles (32 cols)

  // init: h0 = 0, v padding = 0
  for (int i = tid; i < KBH * 512; i += NTHR) hA[i] = 0;
  for (int i = tid; i < KBV * 512; i += NTHR) vA[i] = 0;
  __syncthreads();

  // persistent per-wave state in accumulator layout: cell c, running max, h_new
  v8f cc[JPW], hm[JPW], hnew[JPW];
#pragma unroll
  for (int j = 0; j < JPW; ++j)
#pragma unroll
    for (int r = 0; r < 8; ++r) { cc[j][r] = 0.0f; hm[j][r] = -1e30f; hnew[j][r] = 0.0f; }

  for (int step = 0; step < LL; ++step) {
    // ---------- GEMM1: logits[16,384] = h(bf16) x W_att frags, +b_att ----------
    {
      const u16* wB = opaque(WattF);       // defeat cross-step LICM, keep global AS
      v8f acc;
#pragma unroll
      for (int r = 0; r < 8; ++r) acc[r] = 0.0f;
      for (int kb = 0; kb < KBH; ++kb) {
        v16bf a  = *(const v16bf*)(hA + (kb << 9) + (lane << 4));
        v16bf bb = *(const v16bf*)(wB + ((size_t)(wave * KBH + kb) << 9) + (lane << 4));
        acc = __builtin_amdgcn_wmma_f32_16x16x32_bf16(false, a, false, bb, (short)0, acc, false, false);
      }
      const int Nc = wave * 16 + (lane & 15), Mb = (lane >> 4) << 3;
      float batt_n = opaque(b_att)[Nc];
#pragma unroll
      for (int r = 0; r < 8; ++r)
        att[(Mb + r) * 384 + Nc] = acc[r] + batt_n;
    }
    __syncthreads();

    // ---------- softmax over L=128 for 48 (batch,head) rows ----------
    if (tid < 48) {
      float* row = att + (tid / 3) * 384 + (tid % 3) * 128;
      float mx = row[0];
      for (int l = 1; l < LL; ++l) mx = fmaxf(mx, row[l]);
      float s = 0.0f;
      for (int l = 0; l < LL; ++l) { float e = __expf(row[l] - mx); row[l] = e; s += e; }
      float inv = 1.0f / s;
      for (int l = 0; l < LL; ++l) row[l] *= inv;
    }
    __syncthreads();

    // ---------- v[b,k,d] = sum_l alpha * X  (batched M=3: VALU path) ----------
    for (int o = tid; o < BT * K3D; o += NTHR) {
      int m  = o / K3D, kk = o - m * K3D;          // kk in [0,390)
      int kh = kk / DIN, d = kk - kh * DIN;
      const float* alp = att + m * 384 + kh * 128;
      const float* xr  = X + (size_t)(b0 + m) * LL * DIN + d;
      float acc = 0.0f;
      for (int l = 0; l < LL; ++l) acc = fmaf(alp[l], xr[(size_t)l * DIN], acc);
      vA[a_elem(m, kk)] = f2bf(acc);
    }
    __syncthreads();

    // ---------- GEMM2 + LSTM: two passes, 4 accumulators (i,f,g,o) each ------
#pragma unroll
    for (int jj = 0; jj < JPW; ++jj) {
      const int jt = jbase + jj;                     // this pass' h-column tile
      const u16* wHH = opaque(WhhF);                 // defeat cross-step LICM
      const u16* wIH = opaque(WihF);
      v8f acc[4];
#pragma unroll
      for (int g = 0; g < 4; ++g)
#pragma unroll
        for (int r = 0; r < 8; ++r) acc[g][r] = 0.0f;

      for (int kb = 0; kb < KBH; ++kb) {             // h @ W_hh^T, K=768
        v16bf a = *(const v16bf*)(hA + (kb << 9) + (lane << 4));
        if (kb + 1 < KBH)
          __builtin_prefetch(wHH + ((size_t)(jt * KBH + kb + 1) << 9) + (lane << 4), 0, 0);
#pragma unroll
        for (int g = 0; g < 4; ++g) {
          v16bf bb = *(const v16bf*)(wHH + ((size_t)((g * TPG + jt) * KBH + kb) << 9) + (lane << 4));
          acc[g] = __builtin_amdgcn_wmma_f32_16x16x32_bf16(
              false, a, false, bb, (short)0, acc[g], false, false);
        }
      }
      for (int kb = 0; kb < KBV; ++kb) {             // v @ W_ih^T, K=416 (padded)
        v16bf a = *(const v16bf*)(vA + (kb << 9) + (lane << 4));
        if (kb + 1 < KBV)
          __builtin_prefetch(wIH + ((size_t)(jt * KBV + kb + 1) << 9) + (lane << 4), 0, 0);
#pragma unroll
        for (int g = 0; g < 4; ++g) {
          v16bf bb = *(const v16bf*)(wIH + ((size_t)((g * TPG + jt) * KBV + kb) << 9) + (lane << 4));
          acc[g] = __builtin_amdgcn_wmma_f32_16x16x32_bf16(
              false, a, false, bb, (short)0, acc[g], false, false);
        }
      }

      // LSTM pointwise for this 16x16 tile; defer h write until after barrier
      const float* bc = opaque(bcomb);
      const int ncol = jt * 16 + (lane & 15);
      const float bi = bc[ncol],          bfv = bc[HH + ncol];
      const float bg = bc[2 * HH + ncol], bo  = bc[3 * HH + ncol];
#pragma unroll
      for (int r = 0; r < 8; ++r) {
        float iv = sigmoidf_(acc[0][r] + bi);
        float fv = sigmoidf_(acc[1][r] + bfv);
        float gv = tanhf    (acc[2][r] + bg);
        float ov = sigmoidf_(acc[3][r] + bo);
        float cn = fv * cc[jj][r] + iv * gv;
        cc[jj][r] = cn;
        float hn = ov * tanhf(cn);
        hm[jj][r] = fmaxf(hm[jj][r], hn);
        hnew[jj][r] = hn;
      }
    }
    __syncthreads();   // all hA/vA reads done before h is rewritten

#pragma unroll
    for (int jj = 0; jj < JPW; ++jj) {
      const int ncol = (jbase + jj) * 16 + (lane & 15);
#pragma unroll
      for (int r = 0; r < 8; ++r) {
        int m = r + ((lane >> 4) << 3);
        hA[a_elem(m, ncol)] = f2bf(hnew[jj][r]);
      }
    }
    __syncthreads();
  }

  // ---------------- tail: maxpool -> Dense1 -> AttFeat -> Dense2 -------------
  float* xbuf = (float*)smem;            // [16][768] f32 (aliases hA/vA/att)
  float* ybuf = (float*)(smem + 49152);  // [16][49]
  float* rbuf = (float*)(smem + 52288);  // [2][16][40]
  float* abuf = (float*)(smem + 57408);  // [32]

#pragma unroll
  for (int jj = 0; jj < JPW; ++jj) {
    int ncol = (jbase + jj) * 16 + (lane & 15);
#pragma unroll
    for (int r = 0; r < 8; ++r) {
      int m = r + ((lane >> 4) << 3);
      xbuf[m * HH + ncol] = hm[jj][r];
    }
  }
  __syncthreads();

  for (int o = tid; o < BT * FEAT; o += NTHR) {       // Dense1 + ReLU
    int m = o / FEAT, f = o - m * FEAT;
    const float* wrow = Wd1 + (size_t)f * HH;
    const float* xr   = xbuf + m * HH;
    float a = bd1[f];
    for (int k = 0; k < HH; ++k) a = fmaf(xr[k], wrow[k], a);
    ybuf[m * FEAT + f] = fmaxf(a, 0.0f);
  }
  __syncthreads();

  for (int o = tid; o < 2 * BT * PROJ; o += NTHR) {   // r1 / r2 + leaky_relu
    int s = o / (BT * PROJ);
    int rem = o - s * BT * PROJ;
    int m = rem / PROJ, p = rem - m * PROJ;
    const float* wr = Wr + p * FEAT;
    float a = br[p];
    if (s == 0) {
      for (int f = 0; f < FEAT; ++f) a = fmaf(ybuf[m * FEAT + f], wr[f], a);
    } else {
      const float* fe = Fe + (size_t)(b0 + m) * FEAT;
      for (int f = 0; f < FEAT; ++f) a = fmaf(fe[f], wr[f], a);
    }
    rbuf[(s * BT + m) * PROJ + p] = (a > 0.0f) ? a : 0.01f * a;
  }
  __syncthreads();

  if (tid < 32) {                                     // attention scores
    int s = tid >> 4, m = tid & 15;
    const float* rr = rbuf + (s * BT + m) * PROJ;
    float a = 0.0f;
    for (int p = 0; p < PROJ; ++p) a = fmaf(tanhf(rr[p]), Wa[p], a);
    abuf[s * BT + m] = a;
  }
  __syncthreads();

  if (tid < 32) {                                     // softmax mix + ReLU + Dense2
    int q = tid >> 4, m = tid & 15;
    float a1 = abuf[m], a2 = abuf[BT + m];
    float mx = fmaxf(a1, a2);
    float e1 = __expf(a1 - mx), e2 = __expf(a2 - mx);
    float w1 = e1 / (e1 + e2), w2 = e2 / (e1 + e2);
    const float* r1 = rbuf + m * PROJ;
    const float* r2 = rbuf + (BT + m) * PROJ;
    const float* wd = Wd2 + q * PROJ;
    float a = bd2[q];
    for (int p = 0; p < PROJ; ++p) {
      float sv = fmaxf(w1 * r1[p] + w2 * r2[p], 0.0f);
      a = fmaf(sv, wd[p], a);
    }
    out[(size_t)(b0 + m) * 2 + q] = a;
  }
}

// ---------------------------------------------------------------------------
extern "C" void kernel_launch(void* const* d_in, const int* in_sizes, int n_in,
                              void* d_out, int out_size, void* d_ws, size_t ws_size,
                              hipStream_t stream) {
  const float* X    = (const float*)d_in[0];
  const float* Fe   = (const float*)d_in[1];
  const float* Watt = (const float*)d_in[2];
  const float* batt = (const float*)d_in[3];
  const float* Wih  = (const float*)d_in[4];
  const float* Whh  = (const float*)d_in[5];
  const float* bih  = (const float*)d_in[6];
  const float* bhh  = (const float*)d_in[7];
  const float* Wd1  = (const float*)d_in[8];
  const float* bd1  = (const float*)d_in[9];
  const float* Wr   = (const float*)d_in[10];
  const float* br   = (const float*)d_in[11];
  const float* Wa   = (const float*)d_in[12];
  const float* Wd2  = (const float*)d_in[13];
  const float* bd2  = (const float*)d_in[14];
  float* out = (float*)d_out;

  // workspace layout (bf16 fragment-major weight copies, ~7.5 MB total)
  char* ws = (char*)d_ws;
  u16*   WattF = (u16*)ws;                   //  24*24*512*2 =   589824 B
  u16*   WhhF  = (u16*)(ws + 589824);        // 192*24*512*2 =  4718592 B
  u16*   WihF  = (u16*)(ws + 5308416);       // 192*13*512*2 =  2555904 B
  float* bcomb = (float*)(ws + 7864320);     // 3072*4       =    12288 B

  pack_watt<<<(HH * 384 + 255) / 256, 256, 0, stream>>>(Watt, WattF);
  pack_whh <<<(3072 * HH + 255) / 256, 256, 0, stream>>>(Whh, WhhF);
  pack_wih <<<(3072 * K3DP + 255) / 256, 256, 0, stream>>>(Wih, WihF);
  add_bias <<<(4 * HH + 255) / 256, 256, 0, stream>>>(bih, bhh, bcomb);

  lstm_att_main<<<64, NTHR, 0, stream>>>(X, Fe, batt, WattF, WhhF, WihF, bcomb,
                                         Wd1, bd1, Wr, br, Wa, Wd2, bd2, out);
}
